// QuantBaseWindowAttention_15899968930300
// MI455X (gfx1250) — matrix operand order repaired and checked by reading, not compiled, not run on hardware
//
#include <hip/hip_runtime.h>
#include <hip/hip_bf16.h>

typedef __bf16 bf16_t;
typedef __attribute__((ext_vector_type(16))) __bf16 v16bf;
typedef __attribute__((ext_vector_type(8)))  __bf16 v8bf;
typedef __attribute__((ext_vector_type(4)))  __bf16 v4bf;
typedef __attribute__((ext_vector_type(8)))  float  v8f;
typedef __attribute__((ext_vector_type(4)))  float  v4f;

union V16U { v16bf v; v8bf h[2]; };

__device__ __forceinline__ v8f wmma_bf16(v16bf a, v16bf b, v8f c) {
  // D(f32 16x16) = A(bf16 16x32) * B(bf16 32x16) + C
  return __builtin_amdgcn_wmma_f32_16x16x32_bf16(false, a, false, b, (short)0, c, false, false);
}

__device__ __forceinline__ v8f vzero8() {
  v8f z;
#pragma unroll
  for (int e = 0; e < 8; ++e) z[e] = 0.0f;
  return z;
}

__device__ __forceinline__ v8bf cvt8(const float* __restrict__ p) {
  v4f a = *(const v4f*)p;
  v4f b = *(const v4f*)(p + 4);
  v8bf r;
#pragma unroll
  for (int e = 0; e < 4; ++e) { r[e] = (__bf16)a[e]; r[e + 4] = (__bf16)b[e]; }
  return r;
}

// gfx1250 async global->LDS copy (ASYNCcnt path, CDNA5 ISA).
// VDST VGPR carries the LDS byte address; generic LDS pointers keep it in addr[31:0].
__device__ __forceinline__ void async_b128_to_lds(void* lds_ptr, const void* gptr) {
  unsigned int lds_off = (unsigned int)(unsigned long long)(uintptr_t)lds_ptr;
  unsigned long long ga = (unsigned long long)(uintptr_t)gptr;
  asm volatile("global_load_async_to_lds_b128 %0, %1, off"
               :: "v"(lds_off), "v"(ga)
               : "memory");
}
__device__ __forceinline__ void wait_async0() {
  asm volatile("s_wait_asynccnt 0" ::: "memory");
}

// ---------------------------------------------------------------------------
// Kernel 0: one-shot fp32 -> bf16 conversion (weights), vectorized x4
// ---------------------------------------------------------------------------
__global__ __launch_bounds__(256) void f32_to_bf16_kernel(
    const float* __restrict__ src, bf16_t* __restrict__ dst, int n4) {
  const int i = blockIdx.x * 256 + threadIdx.x;
  if (i < n4) {
    v4f a = *(const v4f*)(src + 4 * i);
    v4bf r;
#pragma unroll
    for (int e = 0; e < 4; ++e) r[e] = (__bf16)a[e];
    *(v4bf*)(dst + 4 * i) = r;
  }
}

// ---------------------------------------------------------------------------
// Kernel 1: qkv = x @ w_qkv^T + b_qkv   (M=131072, K=256, N=768), bf16 out
// Block: 256 threads (8 waves). Tile: 16 rows x 768 cols. Wave w: cols [96w,96w+96).
// B (weights) pre-converted to bf16: fragment loads are pure global_load_b128.
// ---------------------------------------------------------------------------
__global__ __launch_bounds__(256) void qkv_gemm_kernel(
    const float* __restrict__ x, const bf16_t* __restrict__ w_bf,
    const float* __restrict__ b_qkv, bf16_t* __restrict__ qkv) {
  const int tid  = threadIdx.x;
  const int wv   = tid >> 5;
  const int lane = tid & 31;
  const int half = lane >> 4;
  const int idx  = lane & 15;
  const long mBase = (long)blockIdx.x * 16;
  const float* arow = x + (mBase + idx) * 256;

  v8f acc[6];
#pragma unroll
  for (int j = 0; j < 6; ++j) acc[j] = vzero8();

  for (int k0 = 0; k0 < 256; k0 += 32) {
    // A fragment: row idx, K packing 16h..16h+15 (contiguous, matches B)
    V16U a;
    a.h[0] = cvt8(arow + k0 + 16 * half);
    a.h[1] = cvt8(arow + k0 + 16 * half + 8);
    if (k0 + 32 < 256) __builtin_prefetch(arow + k0 + 32, 0, 0);
#pragma unroll
    for (int j = 0; j < 6; ++j) {
      const int n = wv * 96 + j * 16 + idx;        // B column = w_qkv row (contiguous K)
      const bf16_t* brow = w_bf + (long)n * 256 + k0 + 16 * half;
      V16U b;
      b.h[0] = *(const v8bf*)(brow);
      b.h[1] = *(const v8bf*)(brow + 8);
      acc[j] = wmma_bf16(a.v, b.v, acc[j]);
    }
  }

#pragma unroll
  for (int j = 0; j < 6; ++j) {
    const int n = wv * 96 + j * 16 + idx;
    const float bias = b_qkv[n];
#pragma unroll
    for (int r = 0; r < 8; ++r) {
      const long m = mBase + 8 * half + r;          // C layout: VGPR r -> row r+8*half
      qkv[m * 768 + n] = (__bf16)(acc[j][r] + bias);
    }
  }
}

// ---------------------------------------------------------------------------
// Kernel 2: windowed attention. One block (128 thr = 4 waves) per (window, head).
// S = Q K^T * scale + bias; P = softmax(S); O = P V. 64 tokens, head_dim 32.
// Q/K staged via async global->LDS (ASYNCcnt); V transposed through VGPRs.
// ---------------------------------------------------------------------------
__global__ __launch_bounds__(128) void win_attn_kernel(
    const bf16_t* __restrict__ qkv, const float* __restrict__ pos,
    const int* __restrict__ rel, bf16_t* __restrict__ attn_out) {
  __shared__ __align__(16) __bf16 q_s[64][32];
  __shared__ __align__(16) __bf16 k_s[64][32];
  __shared__ __align__(16) __bf16 v_t[32][64];   // transposed: v_t[c][token]
  __shared__ __align__(16) __bf16 p_s[64][64];
  __shared__ __align__(16) float  bias_s[64][64];

  const int tid = threadIdx.x;
  const int wg  = blockIdx.x;
  const int hd  = wg & 7;
  const int win = wg >> 3;
  const int nwI = win & 15;
  const int nhI = (win >> 4) & 15;
  const int bl  = win >> 8;   // combined (b,l) in [0,8)

  // Stage Q, K via async LDS copies; V via VGPRs (transposed store)
  {
    const int t  = tid >> 2;
    const int c0 = (tid & 3) << 3;
    const int wh = t >> 3, ww = t & 7;
    const int y = nhI * 8 + wh, xx = nwI * 8 + ww;
    const long tok = ((long)bl * 128 + y) * 128 + xx;
    const bf16_t* base = qkv + tok * 768 + hd * 32 + c0;
    async_b128_to_lds(&q_s[t][c0], base);
    async_b128_to_lds(&k_s[t][c0], base + 256);
    v8bf vv = *(const v8bf*)(base + 512);
#pragma unroll
    for (int e = 0; e < 8; ++e) v_t[c0 + e][t] = vv[e];
  }

  // Gather relative-position bias table into LDS (overlaps with async copies)
  for (int e = tid; e < 4096; e += 128) {
    const int r0 = rel[2 * e], r1 = rel[2 * e + 1];
    bias_s[e >> 6][e & 63] = pos[r0 * 15 + r1];
  }

  wait_async0();
  __syncthreads();

  const int wv   = tid >> 5;
  const int lane = tid & 31;
  const int half = lane >> 4;
  const int idx  = lane & 15;
  const int iRow = 16 * wv + idx;   // this lane's A-matrix row

  // S = Q K^T : wave wv owns score rows [16wv,16wv+16), 4 column tiles, K=32
  V16U aq;
  aq.h[0] = *(const v8bf*)&q_s[iRow][16 * half];
  aq.h[1] = *(const v8bf*)&q_s[iRow][16 * half + 8];

  v8f s[4];
#pragma unroll
  for (int j = 0; j < 4; ++j) {
    V16U bk;  // B column = K row (contiguous channels); B packing: K=16h..16h+15
    bk.h[0] = *(const v8bf*)&k_s[16 * j + idx][16 * half];
    bk.h[1] = *(const v8bf*)&k_s[16 * j + idx][16 * half + 8];
    s[j] = wmma_bf16(aq.v, bk.v, vzero8());
  }

  // scale + bias + row softmax (row lives in one 16-lane half-group)
  const float SCALE = 0.17677669529663687f;  // 32^-0.5
#pragma unroll
  for (int r = 0; r < 8; ++r) {
    const int row = 16 * wv + 8 * half + r;
    float m = -1e30f;
#pragma unroll
    for (int j = 0; j < 4; ++j) {
      float v = s[j][r] * SCALE + bias_s[row][16 * j + idx];
      s[j][r] = v;
      m = fmaxf(m, v);
    }
#pragma unroll
    for (int msk = 1; msk < 16; msk <<= 1) m = fmaxf(m, __shfl_xor(m, msk, 32));
    float sum = 0.0f;
#pragma unroll
    for (int j = 0; j < 4; ++j) {
      float ev = __expf(s[j][r] - m);
      s[j][r] = ev;
      sum += ev;
    }
#pragma unroll
    for (int msk = 1; msk < 16; msk <<= 1) sum += __shfl_xor(sum, msk, 32);
    const float inv = 1.0f / sum;
#pragma unroll
    for (int j = 0; j < 4; ++j) p_s[row][16 * j + idx] = (__bf16)(s[j][r] * inv);
  }
  __syncthreads();

  // O = P V : rows [16wv,16wv+16) x 32 cols, K=64 in two 32-chunks
  v8f o[2];
  o[0] = vzero8(); o[1] = vzero8();
#pragma unroll
  for (int kc = 0; kc < 2; ++kc) {
    V16U ap;
    ap.h[0] = *(const v8bf*)&p_s[iRow][kc * 32 + 16 * half];
    ap.h[1] = *(const v8bf*)&p_s[iRow][kc * 32 + 16 * half + 8];
#pragma unroll
    for (int ct = 0; ct < 2; ++ct) {
      V16U bv;  // B column c = v_t row (contiguous K)
      bv.h[0] = *(const v8bf*)&v_t[16 * ct + idx][kc * 32 + 16 * half];
      bv.h[1] = *(const v8bf*)&v_t[16 * ct + idx][kc * 32 + 16 * half + 8];
      o[ct] = wmma_bf16(ap.v, bv.v, o[ct]);
    }
  }

  // Scatter O back to (token, m*c) layout as bf16
#pragma unroll
  for (int ct = 0; ct < 2; ++ct) {
    const int c = hd * 32 + 16 * ct + idx;
#pragma unroll
    for (int r = 0; r < 8; ++r) {
      const int row = 16 * wv + 8 * half + r;
      const int wh = row >> 3, ww = row & 7;
      const int y = nhI * 8 + wh, xx = nwI * 8 + ww;
      const long tok = ((long)bl * 128 + y) * 128 + xx;
      attn_out[tok * 256 + c] = (__bf16)o[ct][r];
    }
  }
}

// ---------------------------------------------------------------------------
// Kernel 3: out = attn @ w_out^T + b_out  (M=131072, K=256, N=256), fp32 out
// ---------------------------------------------------------------------------
__global__ __launch_bounds__(256) void out_gemm_kernel(
    const bf16_t* __restrict__ attn, const bf16_t* __restrict__ w_bf,
    const float* __restrict__ b_out, float* __restrict__ out) {
  const int tid  = threadIdx.x;
  const int wv   = tid >> 5;
  const int lane = tid & 31;
  const int half = lane >> 4;
  const int idx  = lane & 15;
  const long mBase = (long)blockIdx.x * 16;
  const bf16_t* arow = attn + (mBase + idx) * 256;

  v8f acc[2];
  acc[0] = vzero8(); acc[1] = vzero8();

  for (int k0 = 0; k0 < 256; k0 += 32) {
    V16U a;   // A already bf16 in workspace
    a.h[0] = *(const v8bf*)(arow + k0 + 16 * half);
    a.h[1] = *(const v8bf*)(arow + k0 + 16 * half + 8);
#pragma unroll
    for (int ct = 0; ct < 2; ++ct) {
      const int n = wv * 32 + ct * 16 + idx;
      const bf16_t* brow = w_bf + (long)n * 256 + k0 + 16 * half;
      V16U b;
      b.h[0] = *(const v8bf*)(brow);
      b.h[1] = *(const v8bf*)(brow + 8);
      acc[ct] = wmma_bf16(a.v, b.v, acc[ct]);
    }
  }

#pragma unroll
  for (int ct = 0; ct < 2; ++ct) {
    const int n = wv * 32 + ct * 16 + idx;
    const float bias = b_out[n];
#pragma unroll
    for (int r = 0; r < 8; ++r) {
      const long m = mBase + 8 * half + r;
      out[m * 256 + n] = acc[ct][r] + bias;
    }
  }
}

// ---------------------------------------------------------------------------
extern "C" void kernel_launch(void* const* d_in, const int* in_sizes, int n_in,
                              void* d_out, int out_size, void* d_ws, size_t ws_size,
                              hipStream_t stream) {
  const float* x     = (const float*)d_in[0];
  const float* w_qkv = (const float*)d_in[1];
  const float* b_qkv = (const float*)d_in[2];
  const float* w_out = (const float*)d_in[3];
  const float* b_out = (const float*)d_in[4];
  const float* pos   = (const float*)d_in[5];
  const int*   rel   = (const int*)d_in[6];

  const long TOKENS = 2L * 4 * 128 * 128;   // 131072
  char* ws = (char*)d_ws;
  bf16_t* qkv_ws  = (bf16_t*)ws;                                  // 201,326,592 B
  bf16_t* attn_ws = (bf16_t*)(ws + (size_t)TOKENS * 768 * 2);     //  67,108,864 B
  bf16_t* wqkv_bf = (bf16_t*)(ws + (size_t)TOKENS * 768 * 2
                                 + (size_t)TOKENS * 256 * 2);     //     393,216 B
  bf16_t* wout_bf = wqkv_bf + 768 * 256;                          //     131,072 B

  // One-shot weight conversion (L2-resident thereafter)
  f32_to_bf16_kernel<<<(768 * 256 / 4 + 255) / 256, 256, 0, stream>>>(w_qkv, wqkv_bf, 768 * 256 / 4);
  f32_to_bf16_kernel<<<(256 * 256 / 4 + 255) / 256, 256, 0, stream>>>(w_out, wout_bf, 256 * 256 / 4);

  qkv_gemm_kernel<<<(int)(TOKENS / 16), 256, 0, stream>>>(x, wqkv_bf, b_qkv, qkv_ws);
  win_attn_kernel<<<16384, 128, 0, stream>>>(qkv_ws, pos, rel, attn_ws);
  out_gemm_kernel<<<(int)(TOKENS / 16), 256, 0, stream>>>(attn_ws, w_out ? wout_bf : wout_bf, b_out, (float*)d_out);
}
